// IMPContext_65824668779068
// MI455X (gfx1250) — matrix operand order repaired and checked by reading, not compiled, not run on hardware
//
#include <hip/hip_runtime.h>
#include <hip/hip_bf16.h>
#include <math.h>

// ---------------- CDNA5 WMMA types ----------------
typedef __bf16 bf16_t;
typedef __attribute__((ext_vector_type(16))) __bf16 v16bf;
typedef __attribute__((ext_vector_type(8)))  __bf16 v8bf;
typedef __attribute__((ext_vector_type(8)))  float  v8f;
typedef __attribute__((ext_vector_type(4)))  float  v4f;
typedef __attribute__((ext_vector_type(2)))  float  v2f;

#define N_OBJ 1024
#define N_REL 4096
#define OBJ_DIM 4096
#define EMBED 200
#define HID 512
#define POOL 4096
#define NCLS 151
#define NHEAD 8
#define DH 64
#define DFF 2048
#define PRE_D (OBJ_DIM + EMBED + 128)        // 4424
#define AUG_D (EMBED + OBJ_DIM + HID)        // 4808

__device__ __forceinline__ unsigned pack_bf16(float x, float y) {
    union { bf16_t h[2]; unsigned u; } p;
    p.h[0] = (bf16_t)x; p.h[1] = (bf16_t)y;
    return p.u;
}
__device__ __forceinline__ v16bf cat8(v8bf a, v8bf b) {
    return __builtin_shufflevector(a, b, 0,1,2,3,4,5,6,7,8,9,10,11,12,13,14,15);
}

// =====================================================================
// Tiled GEMM:  C[M,N] = act(A[M,K] @ W[N,K]^T + bias) (+= optional)
// One wave -> one 16x64 C tile (4 accumulators, A-fragment reuse).
// bf16 operands staged in LDS, fp32 accumulate.
// M multiple of 16, N multiple of 64; K arbitrary even (guarded tail).
// LDS A: [row16][k32] ; LDS B: [n64][k32]
// =====================================================================
__global__ void __launch_bounds__(256)
gemm_wmma_bf16(const float* __restrict__ A, int lda,
               const float* __restrict__ W,           // N x K (row-major)
               const float* __restrict__ bias,
               float* __restrict__ C, int ldc,
               int M, int N, int K, int act, int accum)
{
    __shared__ bf16_t lA[8][16][32];
    __shared__ bf16_t lB[8][64][32];

    const int lane = threadIdx.x & 31;
    const int wv   = threadIdx.x >> 5;
    const int ntn  = N >> 6;                   // 64-wide col tiles
    const int ntiles = (M >> 4) * ntn;

    int t = blockIdx.x * 8 + wv;
    const bool active = (t < ntiles);
    if (!active) t = 0;
    const int tm = t / ntn, tn = t % ntn;
    const int r0 = tm << 4, c0 = tn << 6;

    const int mrow = lane & 15;
    const int hi   = lane >> 4;

    // fill pattern: two lanes per row, each covers 16 k (8 float2)
    const int frow = lane >> 1;                // 0..15
    const int fk0  = (lane & 1) * 16;          // 0 or 16

    v8f acc[4] = {{}, {}, {}, {}};
    const int kfull = K & ~31;

    for (int k0 = 0; k0 < kfull; k0 += 32) {
        // ---- unguarded fill: float2 global loads, packed b32 LDS stores ----
        const float* arow = A + (size_t)(r0 + frow) * lda + k0 + fk0;
        const float* wrow = W + (size_t)(c0 + frow) * (size_t)K + k0 + fk0;
        #pragma unroll
        for (int i = 0; i < 8; ++i) {
            v2f av = *(const v2f*)(arow + 2 * i);
            *(unsigned*)&lA[wv][frow][fk0 + 2 * i] = pack_bf16(av.x, av.y);
        }
        #pragma unroll
        for (int gr = 0; gr < 4; ++gr) {
            const float* wr = wrow + (size_t)(16 * gr) * (size_t)K;
            #pragma unroll
            for (int i = 0; i < 8; ++i) {
                v2f wvv = *(const v2f*)(wr + 2 * i);
                *(unsigned*)&lB[wv][frow + 16 * gr][fk0 + 2 * i] = pack_bf16(wvv.x, wvv.y);
            }
        }
        if (k0 + 32 < K)   // -> global_prefetch_b8
            __builtin_prefetch(arow + 32, 0, 1);
        __syncthreads();

        // ---- fragments via contiguous b128 LDS loads; A reused 4x ----
        v8bf a0 = *(const v8bf*)&lA[wv][mrow][hi * 8];
        v8bf a1 = *(const v8bf*)&lA[wv][mrow][16 + hi * 8];
        v16bf af = cat8(a0, a1);
        #pragma unroll
        for (int ct = 0; ct < 4; ++ct) {
            v8bf b0 = *(const v8bf*)&lB[wv][ct * 16 + mrow][hi * 8];
            v8bf b1 = *(const v8bf*)&lB[wv][ct * 16 + mrow][16 + hi * 8];
            acc[ct] = __builtin_amdgcn_wmma_f32_16x16x32_bf16(
                        false, af, false, cat8(b0, b1), (short)0, acc[ct], false, false);
        }
        __syncthreads();
    }

    if (kfull < K) {   // guarded tail tile (runs at most once)
        #pragma unroll
        for (int i = 0; i < 8; ++i) {
            int kk = fk0 + 2 * i;
            float ax = (kfull + kk < K)     ? A[(size_t)(r0 + frow) * lda + kfull + kk]     : 0.f;
            float ay = (kfull + kk + 1 < K) ? A[(size_t)(r0 + frow) * lda + kfull + kk + 1] : 0.f;
            *(unsigned*)&lA[wv][frow][kk] = pack_bf16(ax, ay);
            #pragma unroll
            for (int gr = 0; gr < 4; ++gr) {
                const float* wr = W + (size_t)(c0 + frow + 16 * gr) * (size_t)K;
                float wx = (kfull + kk < K)     ? wr[kfull + kk]     : 0.f;
                float wy = (kfull + kk + 1 < K) ? wr[kfull + kk + 1] : 0.f;
                *(unsigned*)&lB[wv][frow + 16 * gr][kk] = pack_bf16(wx, wy);
            }
        }
        __syncthreads();
        v8bf a0 = *(const v8bf*)&lA[wv][mrow][hi * 8];
        v8bf a1 = *(const v8bf*)&lA[wv][mrow][16 + hi * 8];
        v16bf af = cat8(a0, a1);
        #pragma unroll
        for (int ct = 0; ct < 4; ++ct) {
            v8bf b0 = *(const v8bf*)&lB[wv][ct * 16 + mrow][hi * 8];
            v8bf b1 = *(const v8bf*)&lB[wv][ct * 16 + mrow][16 + hi * 8];
            acc[ct] = __builtin_amdgcn_wmma_f32_16x16x32_bf16(
                        false, af, false, cat8(b0, b1), (short)0, acc[ct], false, false);
        }
        __syncthreads();
    }

    if (active) {
        #pragma unroll
        for (int ct = 0; ct < 4; ++ct) {
            int col = c0 + ct * 16 + mrow;
            float bv = bias ? bias[col] : 0.f;
            #pragma unroll
            for (int r = 0; r < 8; ++r) {
                int row = r0 + r + 8 * hi;
                float v = acc[ct][r] + bv;
                if (act) v = fmaxf(v, 0.f);
                size_t o = (size_t)row * ldc + col;
                if (accum) C[o] = C[o] + v; else C[o] = v;
            }
        }
    }
}

// =====================================================================
// Flash attention: one wave per (head, 16-query tile). Online softmax
// over 32-key blocks; S and P@V via v_wmma_f32_16x16x32_bf16.
// qkv: [NQ,1536] = [Q|K|V], out o: [NQ,512]
// =====================================================================
__global__ void __launch_bounds__(256)
flash_attn(const float* __restrict__ qkv, float* __restrict__ o, int NQ)
{
    __shared__ float  S[8][16][32];
    __shared__ bf16_t VtT[8][64][32];       // [dim][key] per wave
    __shared__ float  rm[8][16], rl[8][16], rf[8][16];

    const int lane = threadIdx.x & 31;
    const int wv   = threadIdx.x >> 5;
    const int gw   = blockIdx.x * 8 + wv;          // 2048 waves
    const int h    = gw >> 8;                      // head
    const int q0   = (gw & 255) << 4;              // query tile base
    const int mrow = lane & 15;
    const int hi   = lane >> 4;
    const float scale = 0.125f;                    // 1/sqrt(64)

    const float* qrow = qkv + (size_t)(q0 + mrow) * 1536 + h * DH;
    v16bf aq[2];
    #pragma unroll
    for (int kt = 0; kt < 2; ++kt) {
        v4f f0 = *(const v4f*)(qrow + kt * 32 + hi * 8);
        v4f f1 = *(const v4f*)(qrow + kt * 32 + hi * 8 + 4);
        v4f f2 = *(const v4f*)(qrow + kt * 32 + 16 + hi * 8);
        v4f f3 = *(const v4f*)(qrow + kt * 32 + 16 + hi * 8 + 4);
        #pragma unroll
        for (int e = 0; e < 4; ++e) {
            aq[kt][e]      = (bf16_t)f0[e];
            aq[kt][4 + e]  = (bf16_t)f1[e];
            aq[kt][8 + e]  = (bf16_t)f2[e];
            aq[kt][12 + e] = (bf16_t)f3[e];
        }
    }

    if (lane < 16) { rm[wv][lane] = -1e30f; rl[wv][lane] = 0.f; }
    v8f accO[4] = {{}, {}, {}, {}};
    __syncthreads();

    for (int kb = 0; kb < NQ; kb += 32) {
        // ---- stage V block (32 keys x 64 dims) transposed into LDS ----
        {
            const int dpair = lane * 2;   // 0..62
            #pragma unroll 4
            for (int key = 0; key < 32; ++key) {
                v2f vv = *(const v2f*)(qkv + (size_t)(kb + key) * 1536 + 1024 + h * DH + dpair);
                VtT[wv][dpair][key]     = (bf16_t)vv.x;
                VtT[wv][dpair + 1][key] = (bf16_t)vv.y;
            }
        }
        // ---- S tiles for two 16-key groups ----
        #pragma unroll
        for (int g = 0; g < 2; ++g) {
            v8f s = {};
            const float* krow = qkv + (size_t)(kb + g * 16 + mrow) * 1536 + 512 + h * DH;
            #pragma unroll
            for (int kt = 0; kt < 2; ++kt) {
                v4f f0 = *(const v4f*)(krow + kt * 32 + hi * 8);
                v4f f1 = *(const v4f*)(krow + kt * 32 + hi * 8 + 4);
                v4f f2 = *(const v4f*)(krow + kt * 32 + 16 + hi * 8);
                v4f f3 = *(const v4f*)(krow + kt * 32 + 16 + hi * 8 + 4);
                v16bf bk;
                #pragma unroll
                for (int e = 0; e < 4; ++e) {
                    bk[e]      = (bf16_t)f0[e];
                    bk[4 + e]  = (bf16_t)f1[e];
                    bk[8 + e]  = (bf16_t)f2[e];
                    bk[12 + e] = (bf16_t)f3[e];
                }
                s = __builtin_amdgcn_wmma_f32_16x16x32_bf16(
                        false, aq[kt], false, bk, (short)0, s, false, false);
            }
            #pragma unroll
            for (int r = 0; r < 8; ++r)
                S[wv][r + 8 * hi][g * 16 + mrow] = s[r] * scale;
        }
        __syncthreads();

        // ---- per-row online softmax (lanes 0..15 each own one row) ----
        if (lane < 16) {
            float mold = rm[wv][lane];
            float mb = mold;
            #pragma unroll
            for (int c = 0; c < 32; ++c) mb = fmaxf(mb, S[wv][lane][c]);
            float f = __expf(mold - mb);
            float ls = 0.f;
            #pragma unroll
            for (int c = 0; c < 32; ++c) {
                float pv = __expf(S[wv][lane][c] - mb);
                S[wv][lane][c] = pv;
                ls += pv;
            }
            rm[wv][lane] = mb;
            rl[wv][lane] = rl[wv][lane] * f + ls;
            rf[wv][lane] = f;
        }
        __syncthreads();

        // ---- P fragment from LDS ----
        v16bf pf;
        {
            v4f f0 = *(const v4f*)&S[wv][mrow][hi * 8];
            v4f f1 = *(const v4f*)&S[wv][mrow][hi * 8 + 4];
            v4f f2 = *(const v4f*)&S[wv][mrow][16 + hi * 8];
            v4f f3 = *(const v4f*)&S[wv][mrow][16 + hi * 8 + 4];
            #pragma unroll
            for (int e = 0; e < 4; ++e) {
                pf[e]      = (bf16_t)f0[e];
                pf[4 + e]  = (bf16_t)f1[e];
                pf[8 + e]  = (bf16_t)f2[e];
                pf[12 + e] = (bf16_t)f3[e];
            }
        }
        // ---- rescale + accumulate O over 4 dim-tiles ----
        #pragma unroll
        for (int ct = 0; ct < 4; ++ct) {
            v8bf b0 = *(const v8bf*)&VtT[wv][ct * 16 + mrow][hi * 8];
            v8bf b1 = *(const v8bf*)&VtT[wv][ct * 16 + mrow][16 + hi * 8];
            #pragma unroll
            for (int r = 0; r < 8; ++r) accO[ct][r] *= rf[wv][r + 8 * hi];
            accO[ct] = __builtin_amdgcn_wmma_f32_16x16x32_bf16(
                         false, pf, false, cat8(b0, b1), (short)0, accO[ct], false, false);
        }
        __syncthreads();
    }

    #pragma unroll
    for (int ct = 0; ct < 4; ++ct)
        #pragma unroll
        for (int r = 0; r < 8; ++r) {
            int row = r + 8 * hi;
            o[(size_t)(q0 + row) * HID + h * DH + ct * 16 + mrow] =
                accO[ct][r] / rl[wv][row];
        }
}

// =====================================================================
// Per-object prep: softmax@embed, pos-MLP+BN, concatenated buffers
// =====================================================================
__global__ void __launch_bounds__(256)
prep_obj(const float* __restrict__ roi, const float* __restrict__ logits,
         const float* __restrict__ boxes, const int* __restrict__ labels,
         const float* __restrict__ prob_w, const float* __restrict__ label_w,
         const float* __restrict__ fc1w, const float* __restrict__ fc1b,
         const float* __restrict__ bng, const float* __restrict__ bnb,
         const float* __restrict__ bnm, const float* __restrict__ bnv,
         const float* __restrict__ fc2w, const float* __restrict__ fc2b,
         float* __restrict__ pre, float* __restrict__ aug, float* __restrict__ info8)
{
    __shared__ float sprobs[NCLS];
    __shared__ float red[256];
    __shared__ float pbn[32];
    const int i = blockIdx.x, t = threadIdx.x;

    float lmax = -1e30f;
    for (int c = t; c < NCLS; c += 256) lmax = fmaxf(lmax, logits[i * NCLS + c]);
    red[t] = lmax; __syncthreads();
    for (int st = 128; st > 0; st >>= 1) { if (t < st) red[t] = fmaxf(red[t], red[t + st]); __syncthreads(); }
    float mx = red[0]; __syncthreads();
    float ls = 0.f;
    for (int c = t; c < NCLS; c += 256) { float e = __expf(logits[i * NCLS + c] - mx); sprobs[c] = e; ls += e; }
    red[t] = ls; __syncthreads();
    for (int st = 128; st > 0; st >>= 1) { if (t < st) red[t] += red[t + st]; __syncthreads(); }
    const float Ssum = red[0];

    const float x1 = boxes[i * 4 + 0], y1 = boxes[i * 4 + 1];
    const float x2 = boxes[i * 4 + 2], y2 = boxes[i * 4 + 3];
    const float w = x2 - x1, hgt = y2 - y1;
    float info9[9] = {(x1 + x2) * 0.5f, (y1 + y2) * 0.5f, w, hgt, x1, y1, x2, y2, w * hgt};

    if (t < 32) {
        float p = fc1b[t];
        #pragma unroll
        for (int q = 0; q < 9; ++q) p += fc1w[t * 9 + q] * info9[q];
        pbn[t] = bng[t] * (p - bnm[t]) * rsqrtf(bnv[t] + 1e-5f) + bnb[t];
    }
    __syncthreads();

    if (t < 128) {
        float s = fc2b[t];
        #pragma unroll
        for (int q = 0; q < 32; ++q) s += fc2w[t * 32 + q] * pbn[q];
        pre[(size_t)i * PRE_D + OBJ_DIM + EMBED + t] = fmaxf(s, 0.f);
    }
    if (t < EMBED) {
        float d = 0.f;
        for (int c = 0; c < NCLS; ++c) d += sprobs[c] * prob_w[c * EMBED + t];
        pre[(size_t)i * PRE_D + OBJ_DIM + t] = d / Ssum;
        int lbl = labels[i];
        aug[(size_t)i * AUG_D + t] = label_w[lbl * EMBED + t];
    }
    for (int j = t; j < OBJ_DIM; j += 256) {
        float v = roi[(size_t)i * OBJ_DIM + j];
        pre[(size_t)i * PRE_D + j] = v;
        aug[(size_t)i * AUG_D + EMBED + j] = v;
    }
    if (t == 0) {
        float* o8 = info8 + i * 8;
        o8[0] = x1; o8[1] = y1; o8[2] = x2; o8[3] = y2;
        o8[4] = (x1 + x2) * 0.5f; o8[5] = (y1 + y2) * 0.5f; o8[6] = w; o8[7] = hgt;
    }
}

__device__ inline void box_info8(float x1, float y1, float x2, float y2, float* o) {
    o[0] = x1; o[1] = y1; o[2] = x2; o[3] = y2;
    o[4] = (x1 + x2) * 0.5f; o[5] = (y1 + y2) * 0.5f; o[6] = x2 - x1; o[7] = y2 - y1;
}

__global__ void __launch_bounds__(256)
geo_kernel(const float* __restrict__ boxes, const float* __restrict__ info8,
           const int* __restrict__ pairs, float* __restrict__ geo)
{
    int r = blockIdx.x * 256 + threadIdx.x;
    if (r >= N_REL) return;
    int ih = pairs[r * 2], it = pairs[r * 2 + 1];
    float* g = geo + (size_t)r * 32;
    #pragma unroll
    for (int q = 0; q < 8; ++q) { g[q] = info8[ih * 8 + q]; g[8 + q] = info8[it * 8 + q]; }
    float hx1 = boxes[ih * 4], hy1 = boxes[ih * 4 + 1], hx2 = boxes[ih * 4 + 2], hy2 = boxes[ih * 4 + 3];
    float tx1 = boxes[it * 4], ty1 = boxes[it * 4 + 1], tx2 = boxes[it * 4 + 2], ty2 = boxes[it * 4 + 3];
    float ux1 = fminf(hx1, tx1), uy1 = fminf(hy1, ty1), ux2 = fmaxf(hx2, tx2), uy2 = fmaxf(hy2, ty2);
    float ix1 = fmaxf(hx1, tx1), iy1 = fmaxf(hy1, ty1), ix2 = fminf(hx2, tx2), iy2 = fminf(hy2, ty2);
    box_info8(ux1, uy1, ux2, uy2, g + 16);
    float ib[8];
    box_info8(ix1, iy1, ix2, iy2, ib);
    float valid = ((ix2 >= ix1) && (iy2 >= iy1)) ? 1.f : 0.f;
    #pragma unroll
    for (int q = 0; q < 8; ++q) g[24 + q] = ib[q] * valid;
}

__global__ void __launch_bounds__(256)
ctx_kernel(const float* __restrict__ fused, const int* __restrict__ pairs, float* __restrict__ ctx)
{
    size_t idx = (size_t)blockIdx.x * 256 + threadIdx.x;
    if (idx >= (size_t)N_REL * HID) return;
    int r = (int)(idx >> 9), j = (int)(idx & 511);
    ctx[idx] = fused[(size_t)pairs[r * 2] * 1024 + j] *
               fused[(size_t)pairs[r * 2 + 1] * 1024 + 512 + j];
}

__global__ void __launch_bounds__(256)
objpair_kernel(const float* __restrict__ fused, const int* __restrict__ pairs,
               const float* __restrict__ spt, float* __restrict__ op)
{
    size_t idx = (size_t)blockIdx.x * 256 + threadIdx.x;
    if (idx >= (size_t)N_REL * 1024) return;
    int r = (int)(idx >> 10), j = (int)(idx & 1023);
    float v = (j < 512) ? fused[(size_t)pairs[r * 2] * 1024 + j]
                        : fused[(size_t)pairs[r * 2 + 1] * 1024 + 512 + (j - 512)];
    op[idx] = v * spt[idx];
}

__global__ void __launch_bounds__(256)
resid_ln(const float* __restrict__ a, const float* __restrict__ b,
         const float* __restrict__ g, const float* __restrict__ bb,
         float* __restrict__ out, int D)
{
    __shared__ float red[256];
    const int row = blockIdx.x, t = threadIdx.x;
    const float* pa = a + (size_t)row * D;
    const float* pb = b + (size_t)row * D;
    float s = 0.f;
    for (int j = t; j < D; j += 256) s += pa[j] + pb[j];
    red[t] = s; __syncthreads();
    for (int st = 128; st > 0; st >>= 1) { if (t < st) red[t] += red[t + st]; __syncthreads(); }
    float mean = red[0] / D; __syncthreads();
    float v = 0.f;
    for (int j = t; j < D; j += 256) { float d = pa[j] + pb[j] - mean; v += d * d; }
    red[t] = v; __syncthreads();
    for (int st = 128; st > 0; st >>= 1) { if (t < st) red[t] += red[t + st]; __syncthreads(); }
    float rstd = rsqrtf(red[0] / D + 1e-5f);
    for (int j = t; j < D; j += 256)
        out[(size_t)row * D + j] = g[j] * (pa[j] + pb[j] - mean) * rstd + bb[j];
}

__global__ void __launch_bounds__(256)
relu_kernel(const float* __restrict__ a, float* __restrict__ o, size_t n)
{
    size_t i = (size_t)blockIdx.x * 256 + threadIdx.x;
    if (i < n) o[i] = fmaxf(a[i], 0.f);
}

__global__ void __launch_bounds__(256)
copy_kernel(const float* __restrict__ a, float* __restrict__ o, size_t n)
{
    size_t i = (size_t)blockIdx.x * 256 + threadIdx.x;
    if (i < n) o[i] = a[i];
}

// =====================================================================
extern "C" void kernel_launch(void* const* d_in, const int* in_sizes, int n_in,
                              void* d_out, int out_size, void* d_ws, size_t ws_size,
                              hipStream_t stream)
{
    const float* roi      = (const float*)d_in[0];
    const float* unionf   = (const float*)d_in[1];
    const float* logits   = (const float*)d_in[2];
    const float* boxes    = (const float*)d_in[3];
    const int*   labels   = (const int*)d_in[4];
    const int*   pairs    = (const int*)d_in[5];
    const float* prob_w   = (const float*)d_in[6];
    const float* label_w  = (const float*)d_in[7];
    const float* fc1w = (const float*)d_in[8],  *fc1b = (const float*)d_in[9];
    const float* bng  = (const float*)d_in[10], *bnb  = (const float*)d_in[11];
    const float* bnm  = (const float*)d_in[12], *bnv  = (const float*)d_in[13];
    const float* fc2w = (const float*)d_in[14], *fc2b = (const float*)d_in[15];
    const float* objhid_w = (const float*)d_in[16], *objhid_b = (const float*)d_in[17];
    const float* updim_w  = (const float*)d_in[18], *updim_b  = (const float*)d_in[19];
    const float* spt1_w   = (const float*)d_in[20], *spt1_b   = (const float*)d_in[21];
    const float* spt2_w   = (const float*)d_in[22], *spt2_b   = (const float*)d_in[23];
    const float* pairfin_w = (const float*)d_in[24], *pairfin_b = (const float*)d_in[25];
    const float* augfin_w  = (const float*)d_in[26], *augfin_b  = (const float*)d_in[27];
    const float* attn_in_w = (const float*)d_in[28], *attn_in_b = (const float*)d_in[29];
    const float* attn_out_w = (const float*)d_in[30], *attn_out_b = (const float*)d_in[31];
    const float* ln1_g = (const float*)d_in[32], *ln1_b = (const float*)d_in[33];
    const float* ff1_w = (const float*)d_in[34], *ff1_b = (const float*)d_in[35];
    const float* ff2_w = (const float*)d_in[36], *ff2_b = (const float*)d_in[37];
    const float* ln2_g = (const float*)d_in[38], *ln2_b = (const float*)d_in[39];
    const float* mha_fc_w = (const float*)d_in[40], *mha_fc_b = (const float*)d_in[41];

    float* out = (float*)d_out;
    float* rel = out;                                  // [4096,4096]
    float* augfinal = out + (size_t)N_REL * POOL;      // [1024,4096]

    float* ws = (float*)d_ws;
    size_t off = 0;
    float* pre    = ws + off; off += (size_t)N_OBJ * PRE_D;
    float* aug    = ws + off; off += (size_t)N_OBJ * AUG_D;
    float* fusedb = ws + off; off += (size_t)N_OBJ * 1024;
    float* info8  = ws + off; off += (size_t)N_OBJ * 8;
    float* geo    = ws + off; off += (size_t)N_REL * 32;
    float* ctx    = ws + off; off += (size_t)N_REL * HID;
    float* sptmid = ws + off; off += (size_t)N_REL * HID;
    float* spt    = ws + off; off += (size_t)N_REL * 1024;
    float* objpr  = ws + off; off += (size_t)N_REL * 1024;
    float* qkvb   = ws + off; off += (size_t)N_REL * 3 * HID;
    float* attno  = ws + off; off += (size_t)N_REL * HID;
    float* tmp512 = ws + off; off += (size_t)N_REL * HID;
    float* x1b    = ws + off; off += (size_t)N_REL * HID;
    float* ffmid  = ws + off; off += (size_t)N_REL * DFF;
    float* x2b    = ws + off; off += (size_t)N_REL * HID;
    float* xr     = ws + off; off += (size_t)N_REL * HID;

    auto gemm = [&](const float* A, int lda, const float* W, const float* bias,
                    float* C, int ldc, int M, int N, int K, int act, int accum) {
        int ntiles = (M / 16) * (N / 64);
        int blocks = (ntiles + 7) / 8;
        gemm_wmma_bf16<<<blocks, 256, 0, stream>>>(A, lda, W, bias, C, ldc, M, N, K, act, accum);
    };

    prep_obj<<<N_OBJ, 256, 0, stream>>>(roi, logits, boxes, labels, prob_w, label_w,
                                        fc1w, fc1b, bng, bnb, bnm, bnv, fc2w, fc2b,
                                        pre, aug, info8);
    gemm(pre, PRE_D, objhid_w, objhid_b, aug + (EMBED + OBJ_DIM), AUG_D, N_OBJ, HID, PRE_D, 0, 0);
    gemm(aug, AUG_D, updim_w, updim_b, fusedb, 1024, N_OBJ, 1024, AUG_D, 0, 0);
    geo_kernel<<<(N_REL + 255) / 256, 256, 0, stream>>>(boxes, info8, pairs, geo);
    gemm(geo, 32, spt1_w, spt1_b, sptmid, HID, N_REL, HID, 32, 1, 0);
    gemm(sptmid, HID, spt2_w, spt2_b, spt, 1024, N_REL, 1024, HID, 1, 0);
    ctx_kernel<<<(int)(((size_t)N_REL * HID + 255) / 256), 256, 0, stream>>>(fusedb, pairs, ctx);
    objpair_kernel<<<(int)(((size_t)N_REL * 1024 + 255) / 256), 256, 0, stream>>>(fusedb, pairs, spt, objpr);
    gemm(ctx, HID, attn_in_w, attn_in_b, qkvb, 3 * HID, N_REL, 3 * HID, HID, 0, 0);
    flash_attn<<<(NHEAD * (N_REL / 16)) / 8, 256, 0, stream>>>(qkvb, attno, N_REL);
    gemm(attno, HID, attn_out_w, attn_out_b, tmp512, HID, N_REL, HID, HID, 0, 0);
    resid_ln<<<N_REL, 256, 0, stream>>>(ctx, tmp512, ln1_g, ln1_b, x1b, HID);
    gemm(x1b, HID, ff1_w, ff1_b, ffmid, DFF, N_REL, DFF, HID, 1, 0);
    gemm(ffmid, DFF, ff2_w, ff2_b, tmp512, HID, N_REL, HID, DFF, 0, 0);
    resid_ln<<<N_REL, 256, 0, stream>>>(x1b, tmp512, ln2_g, ln2_b, x2b, HID);
    relu_kernel<<<(int)(((size_t)N_REL * HID + 255) / 256), 256, 0, stream>>>(x2b, xr, (size_t)N_REL * HID);
    copy_kernel<<<(int)(((size_t)N_REL * POOL + 255) / 256), 256, 0, stream>>>(unionf, rel, (size_t)N_REL * POOL);
    gemm(xr, HID, mha_fc_w, mha_fc_b, rel, POOL, N_REL, POOL, HID, 1, 1);
    gemm(objpr, 1024, pairfin_w, pairfin_b, rel, POOL, N_REL, POOL, 1024, 1, 1);
    gemm(aug, AUG_D, augfin_w, augfin_b, augfinal, POOL, N_OBJ, POOL, AUG_D, 1, 0);
}
